// SenriStandardAttention_46256797778263
// MI455X (gfx1250) — matrix operand order repaired and checked by reading, not compiled, not run on hardware
//
#include <hip/hip_runtime.h>
#include <hip/hip_bf16.h>
#include <math.h>

// Problem constants (match reference)
#define B_   2
#define S_   2048
#define D_   2048
#define NH_  32
#define NKV_ 8
#define HD_  64

typedef __bf16 bf16_t;
typedef __attribute__((ext_vector_type(16))) __bf16 v16bf;
typedef __attribute__((ext_vector_type(8)))  __bf16 v8bf;
typedef __attribute__((ext_vector_type(8)))  float  v8f;

__device__ __forceinline__ v8f vzero8() {
  v8f z;
#pragma unroll
  for (int i = 0; i < 8; ++i) z[i] = 0.0f;
  return z;
}

__device__ __forceinline__ v8f wmma_bf16(v16bf a, v16bf b, v8f c) {
  // D = A(16x32 bf16) * B(32x16 bf16) + C(16x16 f32)
  return __builtin_amdgcn_wmma_f32_16x16x32_bf16(false, a, false, b, (short)0, c,
                                                 false, false);
}

// A-fragment (16x32 bf16) from row-major [rows x ld] bf16 matrix.
// ISA layout: lanes 0-15 -> M=0..15, K chunks {k0..k0+7, k0+16..k0+23};
//             lanes 16-31 -> same M, K chunks {k0+8.., k0+24..}.
__device__ __forceinline__ v16bf load_a_frag_g(const bf16_t* A, int row0, int ld, int k0) {
  const int lane = threadIdx.x & 31;
  const int m    = lane & 15;
  const int kb   = k0 + ((lane & 16) ? 8 : 0);
  const bf16_t* p = A + (size_t)(row0 + m) * ld + kb;
  v8bf lo = *(const v8bf*)(p);
  v8bf hi = *(const v8bf*)(p + 16);
  v16bf f;
#pragma unroll
  for (int i = 0; i < 8; ++i) { f[i] = lo[i]; f[i + 8] = hi[i]; }
  return f;
}

// B-fragment (32x16 bf16) where B[k][n] = W[n0+n][k] (i.e. we multiply by W^T),
// W row-major [N x ld]. ISA layout: lane n = lane&15; lanes 0-15 hold K chunks
// {k0..k0+7, k0+16..k0+23}; lanes 16-31 hold {k0+8.., k0+24..} (matches A-frag).
__device__ __forceinline__ v16bf load_b_frag_g(const bf16_t* W, int n0, int ld, int k0) {
  const int lane = threadIdx.x & 31;
  const int n    = lane & 15;
  const int kb   = k0 + ((lane & 16) ? 8 : 0);
  const bf16_t* p = W + (size_t)(n0 + n) * ld + kb;
  v8bf lo = *(const v8bf*)(p);
  v8bf hi = *(const v8bf*)(p + 16);
  v16bf f;
#pragma unroll
  for (int i = 0; i < 8; ++i) { f[i] = lo[i]; f[i + 8] = hi[i]; }
  return f;
}

// ---------------------------------------------------------------- convert
__global__ void cvt_f32_bf16_k(const float* __restrict__ in, bf16_t* __restrict__ out,
                               int nelem) {
  int idx = blockIdx.x * blockDim.x + threadIdx.x;
  if (idx < nelem) out[idx] = (bf16_t)in[idx];
}

// ---------------------------------------------------------------- RoPE (in place)
// X layout (B, heads, S, HD) bf16. Pairs (i, i+32), i in [0,32).
__global__ void rope_k(bf16_t* __restrict__ X, const int* __restrict__ pos_ids,
                       int heads) {
  int idx = blockIdx.x * blockDim.x + threadIdx.x;
  int total = B_ * heads * S_ * (HD_ / 2);
  if (idx >= total) return;
  int i  = idx & 31;
  int s  = (idx >> 5) & (S_ - 1);
  int bh = idx >> 16;              // / (32 * 2048)
  int b  = bh / heads;
  float pos  = (float)pos_ids[b * S_ + s];
  float freq = powf(10000.0f, -(float)(2 * i) / (float)HD_);
  float ang  = pos * freq;
  float c = cosf(ang), sn = sinf(ang);
  size_t base = ((size_t)bh * S_ + s) * HD_;
  float x0 = (float)X[base + i];
  float x1 = (float)X[base + i + 32];
  X[base + i]      = (bf16_t)(x0 * c - x1 * sn);
  X[base + i + 32] = (bf16_t)(x1 * c + x0 * sn);
}

// ---------------------------------------------------------------- GEMM (C = A * W^T)
// A: (Mtot x K) bf16 row-major. W: (N x K) bf16 row-major.
// 128 threads = 4 waves; wave w computes a 32(M) x 64(N) tile:
// 2 A-frags x 4 B-frags -> 8 WMMAs per 32-wide k-step (12 b128 loads), so every
// B-fragment is reused across two M strips (1.5 loads/WMMA vs 2.5 before).
// WG tile: 128(M) x 64(N).
// MODE 0: dst bf16 scattered to (B, heads, S, HD)            (Q/K projections)
// MODE 1: dst bf16 packed to (B, heads, HD, S)  [transposed]  (V projection)
// MODE 2: dst f32 row-major (Mtot x N)                        (output projection)
template <int MODE>
__global__ __launch_bounds__(128) void gemm_bf16_k(const bf16_t* __restrict__ A,
                                                   const bf16_t* __restrict__ W,
                                                   void* __restrict__ dst,
                                                   int Mtot, int N, int K, int heads) {
  const int wave    = threadIdx.x >> 5;
  const int lane    = threadIdx.x & 31;
  const int tileN   = blockIdx.x * 64;
  const int rowBase = blockIdx.y * 128 + wave * 32;
  (void)Mtot;

  v8f acc[2][4];
#pragma unroll
  for (int mi2 = 0; mi2 < 2; ++mi2)
#pragma unroll
    for (int j = 0; j < 4; ++j) acc[mi2][j] = vzero8();

  for (int k0 = 0; k0 < K; k0 += 32) {
    v16bf a0 = load_a_frag_g(A, rowBase,      K, k0);
    v16bf a1 = load_a_frag_g(A, rowBase + 16, K, k0);
#pragma unroll
    for (int j = 0; j < 4; ++j) {
      v16bf b = load_b_frag_g(W, tileN + j * 16, K, k0);
      acc[0][j] = wmma_bf16(a0, b, acc[0][j]);
      acc[1][j] = wmma_bf16(a1, b, acc[1][j]);
    }
  }

  const int n   = lane & 15;
  const int hio = (lane & 16) ? 8 : 0;   // C-frag: VGPR r -> row r (lo) or r+8 (hi)

#pragma unroll
  for (int mi2 = 0; mi2 < 2; ++mi2) {
    const int rb = rowBase + mi2 * 16;
    if (MODE == 0) {
      bf16_t* out = (bf16_t*)dst;
#pragma unroll
      for (int j = 0; j < 4; ++j) {
        int col = tileN + j * 16 + n;
        int h = col >> 6, d = col & 63;
#pragma unroll
        for (int r = 0; r < 8; ++r) {
          int row = rb + hio + r;
          int bb = row >> 11;          // / S_
          int s  = row & (S_ - 1);
          out[(((size_t)bb * heads + h) * S_ + s) * HD_ + d] = (bf16_t)acc[mi2][j][r];
        }
      }
    } else if (MODE == 1) {
      bf16_t* out = (bf16_t*)dst;
      int row0 = rb + hio;
      int bb = row0 >> 11;
      int s0 = row0 & (S_ - 1);
#pragma unroll
      for (int j = 0; j < 4; ++j) {
        int col = tileN + j * 16 + n;
        int h = col >> 6, d = col & 63;
        v8bf pk;
#pragma unroll
        for (int r = 0; r < 8; ++r) pk[r] = (bf16_t)acc[mi2][j][r];
        // 8 consecutive-s bf16 -> one 16B store (s0 multiple of 8 -> 16B aligned)
        *(v8bf*)(out + (((size_t)bb * heads + h) * HD_ + d) * S_ + s0) = pk;
      }
    } else {  // MODE 2
      float* out = (float*)dst;
#pragma unroll
      for (int j = 0; j < 4; ++j) {
        int col = tileN + j * 16 + n;
#pragma unroll
        for (int r = 0; r < 8; ++r) {
          int row = rb + hio + r;
          out[(size_t)row * N + col] = acc[mi2][j][r];
        }
      }
    }
  }
}

// ---------------------------------------------------------------- flash attention
// One 64-key block step for a 16-row Q strip. MASKED only for the causal tail
// block; full blocks skip all compare/cndmask work. 16 WMMAs per block; the two
// 4-step width-16 shuffle reductions are amortized over 64 keys (2x fewer
// ds_bpermute and acc-rescale FMACs per key than a 32-key block).
template <bool MASKED>
__device__ __forceinline__ void flash_block(
    int kb0, int q0, int n, int hio, int lane,
    const bf16_t* __restrict__ Kb, const bf16_t* __restrict__ Vb,
    v16bf aq0, v16bf aq1,
    v8f (&accO)[4], float (&mi)[8], float (&li)[8],
    __bf16 (*psh)[64]) {
  const float scale = 0.125f;  // 1/sqrt(HD)

  // ---- scores: 4 x (16x16) tiles over keys [kb0, kb0+64), Kdim = HD = 64
  v8f s[4];
#pragma unroll
  for (int jj = 0; jj < 4; ++jj) {
    v16bf b0 = load_b_frag_g(Kb, kb0 + jj * 16, HD_, 0);
    v16bf b1 = load_b_frag_g(Kb, kb0 + jj * 16, HD_, 32);
    s[jj] = wmma_bf16(aq0, b0, vzero8());
    s[jj] = wmma_bf16(aq1, b1, s[jj]);
  }

  // ---- scale (+ causal mask only on tail); fold 4 frags before shuffling
  float rm[8];
#pragma unroll
  for (int r = 0; r < 8; ++r) {
    int qi = q0 + hio + r;
#pragma unroll
    for (int jj = 0; jj < 4; ++jj) {
      float v = s[jj][r] * scale;
      if (MASKED) v += ((kb0 + jj * 16 + n) <= qi) ? 0.0f : -1e9f;
      s[jj][r] = v;
    }
    rm[r] = fmaxf(fmaxf(s[0][r], s[1][r]), fmaxf(s[2][r], s[3][r]));
  }
#pragma unroll
  for (int off = 8; off >= 1; off >>= 1)
#pragma unroll
    for (int r = 0; r < 8; ++r)
      rm[r] = fmaxf(rm[r], __shfl_xor(rm[r], off, 16));

  // ---- online softmax update
  float rs[8];
#pragma unroll
  for (int r = 0; r < 8; ++r) {
    float mnew = fmaxf(mi[r], rm[r]);
    float corr = __expf(mi[r] - mnew);
    float sum = 0.0f;
#pragma unroll
    for (int jj = 0; jj < 4; ++jj) {
      float p = __expf(s[jj][r] - mnew);
      s[jj][r] = p;
      sum += p;
    }
    rs[r] = sum;
    mi[r] = mnew;
    li[r] *= corr;
#pragma unroll
    for (int j = 0; j < 4; ++j) accO[j][r] *= corr;
  }
#pragma unroll
  for (int off = 8; off >= 1; off >>= 1)
#pragma unroll
    for (int r = 0; r < 8; ++r)
      rs[r] += __shfl_xor(rs[r], off, 16);
#pragma unroll
  for (int r = 0; r < 8; ++r) li[r] += rs[r];

  // ---- P (C-frag layout, 16x64) -> LDS -> two A-fragments
#pragma unroll
  for (int jj = 0; jj < 4; ++jj)
#pragma unroll
    for (int r = 0; r < 8; ++r)
      psh[hio + r][jj * 16 + n] = (bf16_t)s[jj][r];
  asm volatile("s_wait_dscnt 0" ::: "memory");  // per-wave DS RAW fence
  v16bf pa0, pa1;
  {
    const int m  = lane & 15;
    const int kb = (lane & 16) ? 8 : 0;
#pragma unroll
    for (int i = 0; i < 8; ++i) {
      pa0[i]     = psh[m][kb + i];
      pa0[i + 8] = psh[m][kb + 16 + i];
      pa1[i]     = psh[m][32 + kb + i];
      pa1[i + 8] = psh[m][32 + kb + 16 + i];
    }
  }

  // ---- accO += P * V  (V^T rows contiguous in keys -> 32B/lane loads)
#pragma unroll
  for (int dj = 0; dj < 4; ++dj) {
    v16bf vb0 = load_b_frag_g(Vb, dj * 16, S_, kb0);
    v16bf vb1 = load_b_frag_g(Vb, dj * 16, S_, kb0 + 32);
    accO[dj] = wmma_bf16(pa0, vb0, accO[dj]);
    accO[dj] = wmma_bf16(pa1, vb1, accO[dj]);
  }
}

// Q: (B,NH,S,HD) bf16 (RoPE'd). Kc: (B,NKV,S,HD) bf16 (RoPE'd). Vt: (B,NKV,HD,S) bf16.
// O: (B,S,NH*HD) bf16. 128 threads = 4 waves; wave w owns 16 query rows.
__global__ __launch_bounds__(128) void flash_attn_k(const bf16_t* __restrict__ Q,
                                                    const bf16_t* __restrict__ Kc,
                                                    const bf16_t* __restrict__ Vt,
                                                    bf16_t* __restrict__ O) {
  __shared__ __bf16 psh[4][16][64];   // per-wave P-tile staging (C-frag -> A-frag)

  const int wave = threadIdx.x >> 5;
  const int lane = threadIdx.x & 31;
  const int h   = blockIdx.y;
  const int b   = blockIdx.z;
  const int kvh = h / (NH_ / NKV_);
  const int q0  = blockIdx.x * 64 + wave * 16;

  const bf16_t* Qb = Q  + ((size_t)(b * NH_  + h)   * S_)  * HD_;
  const bf16_t* Kb = Kc + ((size_t)(b * NKV_ + kvh) * S_)  * HD_;
  const bf16_t* Vb = Vt + ((size_t)(b * NKV_ + kvh) * HD_) * S_;

  // Q strip 16x64 kept resident as two A-fragments
  v16bf aq0 = load_a_frag_g(Qb, q0, HD_, 0);
  v16bf aq1 = load_a_frag_g(Qb, q0, HD_, 32);

  v8f accO[4];
#pragma unroll
  for (int j = 0; j < 4; ++j) accO[j] = vzero8();
  float mi[8], li[8];
#pragma unroll
  for (int r = 0; r < 8; ++r) { mi[r] = -1e30f; li[r] = 0.0f; }

  const int n   = lane & 15;
  const int hio = (lane & 16) ? 8 : 0;

  // Blocks [kb0, kb0+64) are fully unmasked when kb0+63 <= q0 (min row of strip).
  const int kend       = q0 + 16;                                   // causal bound
  const int fullBlocks = (q0 >= 63) ? (((q0 - 63) >> 6) + 1) : 0;
  const int kbFullEnd  = fullBlocks << 6;

  int kb0 = 0;
  for (; kb0 < kbFullEnd; kb0 += 64)
    flash_block<false>(kb0, q0, n, hio, lane, Kb, Vb, aq0, aq1, accO, mi, li, psh[wave]);
  for (; kb0 < kend; kb0 += 64)
    flash_block<true>(kb0, q0, n, hio, lane, Kb, Vb, aq0, aq1, accO, mi, li, psh[wave]);

  // ---- epilogue: O[b, qi, h*64 + d] = acc / l
#pragma unroll
  for (int j = 0; j < 4; ++j) {
#pragma unroll
    for (int r = 0; r < 8; ++r) {
      int qi = q0 + hio + r;
      int d  = j * 16 + n;
      O[((size_t)b * S_ + qi) * (NH_ * HD_) + h * HD_ + d] =
          (bf16_t)(accO[j][r] / li[r]);
    }
  }
}

// ---------------------------------------------------------------- launcher
extern "C" void kernel_launch(void* const* d_in, const int* in_sizes, int n_in,
                              void* d_out, int out_size, void* d_ws, size_t ws_size,
                              hipStream_t stream) {
  (void)in_sizes; (void)n_in; (void)out_size; (void)ws_size;
  const float* hs = (const float*)d_in[0];
  /* d_in[1] attention_mask unused: causal 0/-1e9 computed inline (identical) */
  const int*   pos = (const int*)d_in[2];
  const float* wq = (const float*)d_in[3];
  const float* wk = (const float*)d_in[4];
  const float* wv = (const float*)d_in[5];
  const float* wo = (const float*)d_in[6];

  char* w = (char*)d_ws;
  bf16_t* hs_bf = (bf16_t*)w; w += (size_t)B_ * S_ * D_ * 2;
  bf16_t* wq_bf = (bf16_t*)w; w += (size_t)NH_ * HD_ * D_ * 2;
  bf16_t* wk_bf = (bf16_t*)w; w += (size_t)NKV_ * HD_ * D_ * 2;
  bf16_t* wv_bf = (bf16_t*)w; w += (size_t)NKV_ * HD_ * D_ * 2;
  bf16_t* wo_bf = (bf16_t*)w; w += (size_t)D_ * NH_ * HD_ * 2;
  bf16_t* q_bf  = (bf16_t*)w; w += (size_t)B_ * NH_ * S_ * HD_ * 2;
  bf16_t* k_bf  = (bf16_t*)w; w += (size_t)B_ * NKV_ * S_ * HD_ * 2;
  bf16_t* vT_bf = (bf16_t*)w; w += (size_t)B_ * NKV_ * HD_ * S_ * 2;
  bf16_t* at_bf = (bf16_t*)w; w += (size_t)B_ * S_ * NH_ * HD_ * 2;

  // 1) fp32 -> bf16 (L2-resident, ~75MB total traffic, negligible vs GEMMs)
  auto cvt = [&](const float* src, bf16_t* dst, size_t nel) {
    cvt_f32_bf16_k<<<dim3((unsigned)((nel + 255) / 256)), dim3(256), 0, stream>>>(
        src, dst, (int)nel);
  };
  cvt(hs, hs_bf, (size_t)B_ * S_ * D_);
  cvt(wq, wq_bf, (size_t)NH_ * HD_ * D_);
  cvt(wk, wk_bf, (size_t)NKV_ * HD_ * D_);
  cvt(wv, wv_bf, (size_t)NKV_ * HD_ * D_);
  cvt(wo, wo_bf, (size_t)D_ * NH_ * HD_);

  const int Mtot = B_ * S_;
  // 2) QKV projections (WMMA)
  gemm_bf16_k<0><<<dim3((NH_ * HD_) / 64, Mtot / 128), 128, 0, stream>>>(
      hs_bf, wq_bf, q_bf, Mtot, NH_ * HD_, D_, NH_);
  gemm_bf16_k<0><<<dim3((NKV_ * HD_) / 64, Mtot / 128), 128, 0, stream>>>(
      hs_bf, wk_bf, k_bf, Mtot, NKV_ * HD_, D_, NKV_);
  gemm_bf16_k<1><<<dim3((NKV_ * HD_) / 64, Mtot / 128), 128, 0, stream>>>(
      hs_bf, wv_bf, vT_bf, Mtot, NKV_ * HD_, D_, NKV_);

  // 3) RoPE on Q and K
  rope_k<<<dim3((B_ * NH_ * S_ * 32 + 255) / 256), 256, 0, stream>>>(q_bf, pos, NH_);
  rope_k<<<dim3((B_ * NKV_ * S_ * 32 + 255) / 256), 256, 0, stream>>>(k_bf, pos, NKV_);

  // 4) flash attention (WMMA scores + WMMA P*V, online softmax, causal tail split)
  flash_attn_k<<<dim3(S_ / 64, NH_, B_), 128, 0, stream>>>(q_bf, k_bf, vT_bf, at_bf);

  // 5) output projection -> fp32 d_out
  gemm_bf16_k<2><<<dim3(D_ / 64, Mtot / 128), 128, 0, stream>>>(
      at_bf, wo_bf, d_out, Mtot, D_, NH_ * HD_, 0);
}
